// Model_56358560858161
// MI455X (gfx1250) — matrix operand order repaired and checked by reading, not compiled, not run on hardware
//
#include <hip/hip_runtime.h>
#include <hip/hip_bf16.h>

// ---------------- problem constants (from reference) ----------------
#define SEQ_Q   4096
#define SEQ_K   4096
#define HDIM    128
#define BQ      64            // q rows per workgroup (4 waves x 16)
#define BK      64            // kv columns per tile
#define NWAVE   4
#define NTHREADS 128
#define SCALE    11.313708498984761f       // sqrt(128)
#define QSCALE   16.32129585515648f        // sqrt(128) * log2(e)  (exp2 domain)
#define INV_KEEP (1.0f / 0.9f)             // inverted dropout, p = 0.1

typedef __attribute__((ext_vector_type(16))) __bf16 v16bf;
typedef __attribute__((ext_vector_type(8)))  __bf16 v8bf;
typedef __attribute__((ext_vector_type(8)))  float  v8f;
typedef __attribute__((ext_vector_type(4)))  float  v4f;

// ---------------- gfx1250 async global->LDS copy (guarded) ----------------
#if defined(__has_builtin)
#  if __has_builtin(__builtin_amdgcn_global_load_async_to_lds_b128)
#    define HAVE_ASYNC 1
#  endif
#endif
#ifndef HAVE_ASYNC
#  define HAVE_ASYNC 0
#endif

// builtin prototype (from hipcc diagnostic): param0 = global int4*, so:
typedef int v4i __attribute__((vector_size(4 * sizeof(int))));
typedef __attribute__((address_space(1))) v4i g_v4i;   // prints as "__device__"
typedef __attribute__((address_space(3))) v4i l_v4i;   // prints as "__shared__"

static __device__ inline void cp_b128(void* lds_dst, const void* gsrc) {
#if HAVE_ASYNC
    __builtin_amdgcn_global_load_async_to_lds_b128(
        (g_v4i*)gsrc, (l_v4i*)lds_dst, /*offset=*/0, /*cpol=*/0);
#else
    *(uint4*)lds_dst = *(const uint4*)gsrc;   // global_load_b128 + ds_store_b128
#endif
}

#if HAVE_ASYNC
#  if __has_builtin(__builtin_amdgcn_s_wait_asynccnt)
#    define WAIT_ASYNC(n) __builtin_amdgcn_s_wait_asynccnt(n)
#  else
#    define WAIT_ASYNC(n) asm volatile("s_wait_asynccnt %0" :: "n"(n) : "memory")
#  endif
#else
#  define WAIT_ASYNC(n) ((void)0)
#endif

// ---------------- helpers ----------------
static __device__ inline v16bf combine16(v8bf lo, v8bf hi) {
    return __builtin_shufflevector(lo, hi, 0,1,2,3,4,5,6,7,8,9,10,11,12,13,14,15);
}

static __device__ inline v8bf load8_cvt(const float* __restrict__ g) {
    v4f a = *(const v4f*)(g);
    v4f b = *(const v4f*)(g + 4);
    v8bf r;
#pragma unroll
    for (int i = 0; i < 4; ++i) { r[i] = (__bf16)a[i]; r[i + 4] = (__bf16)b[i]; }
    return r;
}

static __device__ inline v8f wmma_bf16(v16bf a, v16bf b, v8f c) {
    return __builtin_amdgcn_wmma_f32_16x16x32_bf16(
        false, a, false, b, (short)0, c, false, false);
}

static __device__ inline float fast_exp2(float x) {
#if defined(__has_builtin)
#  if __has_builtin(__builtin_amdgcn_exp2f)
    return __builtin_amdgcn_exp2f(x);
#  else
    return exp2f(x);
#  endif
#else
    return exp2f(x);
#endif
}

// ---------------- pre-pass: K fp32 -> bf16 (same layout) ----------------
__global__ void cvt_k_bf16(const float* __restrict__ k, __bf16* __restrict__ out) {
    const int idx = blockIdx.x * 256 + threadIdx.x;      // float4 index
    const size_t e0 = (size_t)idx * 4;
    v4f x = *(const v4f*)(k + e0);
#pragma unroll
    for (int j = 0; j < 4; ++j) out[e0 + j] = (__bf16)x[j];
}

// ---------------- pre-pass: V fp32 -> bf16, transposed per batch ----------------
// out layout: [b][h][kv]
__global__ void cvt_v_bf16_t(const float* __restrict__ v, __bf16* __restrict__ out) {
    const int idx = blockIdx.x * 256 + threadIdx.x;      // float4 index
    const size_t e0 = (size_t)idx * 4;
    const int b  = (int)(e0 / ((size_t)SEQ_K * HDIM));
    const int rm = (int)(e0 % ((size_t)SEQ_K * HDIM));
    const int kv = rm / HDIM;
    const int h  = rm % HDIM;
    v4f x = *(const v4f*)(v + e0);
    __bf16* ob = out + (size_t)b * SEQ_K * HDIM;
#pragma unroll
    for (int j = 0; j < 4; ++j) ob[(size_t)(h + j) * SEQ_K + kv] = (__bf16)x[j];
}

// ---------------- main flash-attention kernel ----------------
__global__ __launch_bounds__(NTHREADS)
void fattn_fwd_bf16_wmma(const float* __restrict__ q,
                         const __bf16* __restrict__ kbf,   // [b][kv][h] bf16
                         const __bf16* __restrict__ vtbf,  // [b][h][kv] bf16
                         float* __restrict__ out)
{
    const int tid  = threadIdx.x;
    const int lane = tid & 31;
    const int wave = tid >> 5;
    const int half = lane >> 4;
    const int l15  = lane & 15;

    const int b  = blockIdx.y;
    const int q0 = blockIdx.x * BQ + wave * 16;

    const float*  qb  = q    + (size_t)b * SEQ_Q * HDIM;
    const __bf16* kb  = kbf  + (size_t)b * SEQ_K * HDIM;
    const __bf16* vtb = vtbf + (size_t)b * SEQ_K * HDIM;
    float*        ob  = out  + (size_t)b * SEQ_Q * HDIM;

    // double-buffered bf16 tiles + per-wave P patch
    __shared__ __bf16 sK [2][BK * HDIM];     // 2 x 16 KB
    __shared__ __bf16 sVt[2][HDIM * BK];     // 2 x 16 KB
    __shared__ __bf16 sP [NWAVE * 16 * BK];  //     8 KB

    // ---- Q A-fragments (fp32 -> bf16 once) ----
    v16bf qa[4];
    {
        const float* qrow = qb + (size_t)(q0 + l15) * HDIM;
#pragma unroll
        for (int f = 0; f < 4; ++f) {
            const int kb0 = f * 32 + half * 8;
            qa[f] = combine16(load8_cvt(qrow + kb0), load8_cvt(qrow + kb0 + 16));
        }
    }

    v8f oacc[8];
#pragma unroll
    for (int f = 0; f < 8; ++f)
#pragma unroll
        for (int r = 0; r < 8; ++r) oacc[f][r] = 0.0f;

    float mrow[8], lrow[8];
#pragma unroll
    for (int r = 0; r < 8; ++r) { mrow[r] = -__builtin_inff(); lrow[r] = 0.0f; }

    __bf16* pw = &sP[wave * 16 * BK];

    // tile copy: 16 x b128 per thread (8 K + 8 Vt) => asynccnt 16 per wave per tile
    auto issue_tile = [&](int kv0, int buf) {
        const char* gk = (const char*)(kb + (size_t)kv0 * HDIM);   // contiguous 16 KB
        char*       lk = (char*)&sK[buf][0];
#pragma unroll
        for (int t = 0; t < 8; ++t) {
            const int off = (t * NTHREADS + tid) * 16;
            cp_b128(lk + off, gk + off);
        }
        const char* gv = (const char*)vtb + (size_t)kv0 * 2;       // rows of 128 B
        char*       lv = (char*)&sVt[buf][0];
#pragma unroll
        for (int t = 0; t < 8; ++t) {
            const int idx = t * NTHREADS + tid;
            const int row = idx >> 3;
            const int seg = (idx & 7) * 16;
            cp_b128(lv + row * (BK * 2) + seg,
                    gv + (size_t)row * (SEQ_K * 2) + seg);
        }
    };

    const int NT = SEQ_K / BK;
    issue_tile(0, 0);

    for (int it = 0; it < NT; ++it) {
        const int cur = it & 1;
        const int kv0 = it * BK;

        if (it + 1 < NT) {
            issue_tile(kv0 + BK, cur ^ 1);
            WAIT_ASYNC(16);        // tile `it` complete (in-order), tile it+1 in flight
        } else {
            WAIT_ASYNC(0);
        }
        __syncthreads();           // publish tile `it` to all waves

        // ---- S = Q * K^T (16 x 64 per wave), scaled into exp2 domain ----
        v8f sfr[4];
#pragma unroll
        for (int n = 0; n < 4; ++n) {
            v8f c;
#pragma unroll
            for (int r = 0; r < 8; ++r) c[r] = 0.0f;
            const __bf16* krow = &sK[cur][(n * 16 + l15) * HDIM + half * 8];
#pragma unroll
            for (int ks = 0; ks < 4; ++ks) {
                v8bf lo = *(const v8bf*)(krow + ks * 32);
                v8bf hi = *(const v8bf*)(krow + ks * 32 + 16);
                c = wmma_bf16(qa[ks], combine16(lo, hi), c);
            }
#pragma unroll
            for (int r = 0; r < 8; ++r) c[r] *= QSCALE;
            sfr[n] = c;
        }

        // ---- row max (row = half*8 + r, col = n*16 + l15) ----
        float mnew[8];
        int changed = 0;
#pragma unroll
        for (int r = 0; r < 8; ++r) {
            float mx = fmaxf(fmaxf(sfr[0][r], sfr[1][r]), fmaxf(sfr[2][r], sfr[3][r]));
#pragma unroll
            for (int s = 8; s >= 1; s >>= 1)
                mx = fmaxf(mx, __shfl_xor(mx, s, 32));
            changed |= (mx > mrow[r]);
            mnew[r] = fmaxf(mrow[r], mx);
        }

        // ---- p = exp2(s - m), dropout to P (bf16), row sums ----
        float rs[8];
#pragma unroll
        for (int r = 0; r < 8; ++r) {
            float acc = 0.0f;
            const int qrow = q0 + half * 8 + r;
#pragma unroll
            for (int n = 0; n < 4; ++n) {
                const float p = fast_exp2(sfr[n][r] - mnew[r]);
                acc += p;                       // denominator pre-dropout
                const int kcol = kv0 + n * 16 + l15;
                unsigned u = (unsigned)(qrow * SEQ_K + kcol) + (unsigned)b * 0x1000000u;
                unsigned h = u * 2654435761u;  h ^= h >> 16;
                const float pd = ((h & 0xFFFFu) < 6554u) ? 0.0f : p * INV_KEEP;
                pw[(half * 8 + r) * BK + n * 16 + l15] = (__bf16)pd;
            }
#pragma unroll
            for (int s = 8; s >= 1; s >>= 1) acc += __shfl_xor(acc, s, 32);
            rs[r] = acc;
        }

        // ---- O / l correction only when some row max moved ----
        if (__any(changed)) {
#pragma unroll
            for (int r = 0; r < 8; ++r) {
                const float c = fast_exp2(mrow[r] - mnew[r]);
                lrow[r] = lrow[r] * c + rs[r];
                mrow[r] = mnew[r];
#pragma unroll
                for (int f = 0; f < 8; ++f) oacc[f][r] *= c;
            }
        } else {
#pragma unroll
            for (int r = 0; r < 8; ++r) lrow[r] += rs[r];
        }

        // ---- reload P in A-layout ----
        v16bf pa[2];
        {
            const __bf16* prow = pw + l15 * BK;
#pragma unroll
            for (int f = 0; f < 2; ++f) {
                const int kb0 = f * 32 + half * 8;
                v8bf lo = *(const v8bf*)(prow + kb0);
                v8bf hi = *(const v8bf*)(prow + kb0 + 16);
                pa[f] = combine16(lo, hi);
            }
        }

        // ---- O += P * V ----
#pragma unroll
        for (int f = 0; f < 8; ++f) {
            v8f c = oacc[f];
            const __bf16* vrow = &sVt[cur][(f * 16 + l15) * BK + half * 8];
#pragma unroll
            for (int ks = 0; ks < 2; ++ks) {
                v8bf lo = *(const v8bf*)(vrow + ks * 32);
                v8bf hi = *(const v8bf*)(vrow + ks * 32 + 16);
                c = wmma_bf16(pa[ks], combine16(lo, hi), c);
            }
            oacc[f] = c;
        }

        __syncthreads();   // everyone done with buffer `cur` before it is refilled
    }

    // ---- epilogue ----
#pragma unroll
    for (int r = 0; r < 8; ++r) {
        const float inv = 1.0f / lrow[r];
        float* orow = ob + (size_t)(q0 + half * 8 + r) * HDIM;
#pragma unroll
        for (int f = 0; f < 8; ++f)
            orow[f * 16 + l15] = oacc[f][r] * inv;
    }
}

// ---------------- launcher ----------------
extern "C" void kernel_launch(void* const* d_in, const int* in_sizes, int n_in,
                              void* d_out, int out_size, void* d_ws, size_t ws_size,
                              hipStream_t stream) {
    const float* q = (const float*)d_in[0];
    const float* k = (const float*)d_in[1];
    const float* v = (const float*)d_in[2];
    float* out = (float*)d_out;

    const int batches = in_sizes[0] / (SEQ_Q * HDIM);   // = 4
    const size_t kv_elems = (size_t)batches * SEQ_K * HDIM;

    __bf16* wsK  = (__bf16*)d_ws;               // bf16 K, [b][kv][h]
    __bf16* wsVt = wsK + kv_elems;              // bf16 V^T, [b][h][kv]

    {   // pre-pass conversions (bandwidth-bound, run once per call)
        const int n4 = (int)(kv_elems / 4);
        dim3 g(n4 / 256), blk(256);
        cvt_k_bf16  <<<g, blk, 0, stream>>>(k, wsK);
        cvt_v_bf16_t<<<g, blk, 0, stream>>>(v, wsVt);
    }

    dim3 grid(SEQ_Q / BQ, batches);
    dim3 block(NTHREADS);
    fattn_fwd_bf16_wmma<<<grid, block, 0, stream>>>(q, wsK, wsVt, out);
}
